// LucyRNNCellTriton_63316407877922
// MI455X (gfx1250) — compile-verified, matrix operand order checked by reading, not run on hardware
//
#include <hip/hip_runtime.h>
#include <hip/hip_bf16.h>
#include <cstdint>

// Problem constants (from reference: B=8, T=2048, D=1024)
#define BB   8
#define TT   2048
#define DD   1024
#define KK   DD            // GEMM K
#define MM   (BB * TT)     // GEMM M = 16384
#define NG   6             // gates

typedef __attribute__((ext_vector_type(16))) __bf16 v16bf;
typedef __attribute__((ext_vector_type(8)))  float  v8f;

// ---------------------------------------------------------------------------
// f32 -> bf16 (round-to-nearest-even) pack kernel
// ---------------------------------------------------------------------------
__global__ void pack_bf16_kernel(const float* __restrict__ src,
                                 unsigned short* __restrict__ dst, int n) {
    int i = blockIdx.x * blockDim.x + threadIdx.x;
    int stride = gridDim.x * blockDim.x;
    for (; i < n; i += stride) {
        unsigned int u = __float_as_uint(src[i]);
        unsigned int r = (u + 0x7FFFu + ((u >> 16) & 1u)) >> 16;
        dst[i] = (unsigned short)r;
    }
}

// ---------------------------------------------------------------------------
// Load one 16x32 bf16 WMMA fragment from row-major memory (row stride = KK).
// CDNA5 16-bit A layout (CDNA5 ISA): lane m in 0..15 holds row m with
// K = {k0..k0+7, k0+16..k0+23}; lanes 16..31 hold row m with
// K = {k0+8..k0+15, k0+24..k0+31}.  B (K x 16) from W^T uses the same per-lane
// pattern with lane = column n reading W row o contiguously along K.
// ---------------------------------------------------------------------------
struct bf16x16_bits { uint4 lo, hi; };

__device__ __forceinline__ v16bf load_frag16x32(const unsigned short* __restrict__ base,
                                                int row0, int k0, int lane) {
    int m = lane & 15;
    int g = lane >> 4;                 // 0 or 1
    const unsigned short* p = base + (size_t)(row0 + m) * KK + k0 + g * 8;
    bf16x16_bits t;
    t.lo = *(const uint4*)(p);         // 8 bf16: K = k0 + g*8 + 0..7
    t.hi = *(const uint4*)(p + 16);    // 8 bf16: K = k0 + 16 + g*8 + 0..7
    return __builtin_bit_cast(v16bf, t);
}

__device__ __forceinline__ float fast_sigmoid(float x) {
    return 1.0f / (1.0f + __expf(-x));
}

// ---------------------------------------------------------------------------
// Fused GEMM (bf16 WMMA, f32 accum) + bias + RMS over gates + pointwise gate
// math.  Each wave: 32 M-rows (2 A-frags) x 16 d-channels, all 6 gates
// (12 v8f accums).  Two A fragments share every B fragment -> 16 b128 loads
// per 12 WMMAs per K-step (0.021 B/MAC), halving W-side L2 traffic vs a
// 16-row tile.  Writes interleaved float4 {z, kv, h_pre, decay} per (row, d).
// ---------------------------------------------------------------------------
__global__ void __launch_bounds__(256)
gates_gemm_kernel(const unsigned short* __restrict__ xb,   // [MM, KK] bf16
                  const unsigned short* __restrict__ wb,   // [6*DD, KK] bf16
                  const float* __restrict__ bias,          // [6*DD]
                  float4* __restrict__ scanbuf) {          // [MM, DD]
    const int lane = threadIdx.x & 31;
    const int wave = threadIdx.x >> 5;
    const int d0 = blockIdx.x * 16;               // channel tile
    const int m0 = blockIdx.y * 256 + wave * 32;  // row tile (per wave, 32 rows)

    v8f acc[2][NG];
#pragma unroll
    for (int f = 0; f < 2; ++f)
#pragma unroll
        for (int g = 0; g < NG; ++g)
            acc[f][g] = (v8f){0.f,0.f,0.f,0.f,0.f,0.f,0.f,0.f};

    for (int kb = 0; kb < KK; kb += 32) {
        v16bf a0 = load_frag16x32(xb, m0,      kb, lane);
        v16bf a1 = load_frag16x32(xb, m0 + 16, kb, lane);
#pragma unroll
        for (int g = 0; g < NG; ++g) {
            v16bf bm = load_frag16x32(wb, g * DD + d0, kb, lane);
            acc[0][g] = __builtin_amdgcn_wmma_f32_16x16x32_bf16(
                false, a0, false, bm, (short)0, acc[0][g], false, false);
            acc[1][g] = __builtin_amdgcn_wmma_f32_16x16x32_bf16(
                false, a1, false, bm, (short)0, acc[1][g], false, false);
        }
    }

    // C/D layout: VGPR r, lanes 0-15 -> (M=r, N=lane); lanes 16-31 -> (M=8+r).
    const int n  = lane & 15;
    const int mh = (lane >> 4) * 8;

    float bsc[NG];
#pragma unroll
    for (int g = 0; g < NG; ++g) bsc[g] = bias[g * DD + d0 + n];

#pragma unroll
    for (int f = 0; f < 2; ++f) {
#pragma unroll
        for (int r = 0; r < 8; ++r) {
            float gg[NG];
            float ss = 0.f;
#pragma unroll
            for (int g = 0; g < NG; ++g) {
                gg[g] = acc[f][g][r] + bsc[g];
                ss += gg[g] * gg[g];
            }
            float inv_rms = __frsqrt_rn(ss * (1.0f / 6.0f) + 1e-6f);

            float z   = fast_sigmoid(gg[1] * inv_rms);
            float kv  = (gg[2] * inv_rms) * (gg[3] * inv_rms);
            float hp  = gg[4] * inv_rms;
            float dec = fast_sigmoid(gg[5] * inv_rms);

            size_t row = (size_t)(m0 + f * 16 + mh + r);
            scanbuf[row * DD + d0 + n] = make_float4(z, kv, hp, dec);
        }
    }
}

// ---------------------------------------------------------------------------
// Sequential scan over T per (b, d) chain.  8192 independent chains.
// One coalesced float4 load per step; prefetch a few steps ahead.
// ---------------------------------------------------------------------------
__global__ void __launch_bounds__(256)
scan_kernel(const float* __restrict__ h0, const float* __restrict__ s0,
            const float4* __restrict__ scanbuf,
            float* __restrict__ out, float* __restrict__ s_out) {
    int idx = blockIdx.x * blockDim.x + threadIdx.x;   // 0 .. B*D-1
    int b = idx >> 10;            // / DD
    int d = idx & (DD - 1);       // % DD

    float h = h0[idx];
    float s = s0[idx];

    const float4* p = scanbuf + (size_t)b * TT * DD + d;
    float*        o = out     + (size_t)b * TT * DD + d;

    for (int t = 0; t < TT; ++t) {
        __builtin_prefetch(p + (size_t)(t + 8) * DD, 0, 0);  // global_prefetch
        float4 v = p[(size_t)t * DD];                        // {z, kv, hp, dec}
        s = v.w * s + v.y;
        float c = 2.0f * fast_sigmoid(2.0f * (v.z + s)) - 1.0f;  // tanh(hp+s)
        h = (1.0f - v.x) * c + v.x * h;
        o[(size_t)t * DD] = h;
    }
    s_out[idx] = s;
}

// ---------------------------------------------------------------------------
// Launch: pack x/W to bf16 in ws, fused WMMA GEMM+gates, then scan.
// ws layout: [x_bf16 32MB][w_bf16 12MB][align 256][scanbuf 256MB]
// ---------------------------------------------------------------------------
extern "C" void kernel_launch(void* const* d_in, const int* in_sizes, int n_in,
                              void* d_out, int out_size, void* d_ws, size_t ws_size,
                              hipStream_t stream) {
    const float* x  = (const float*)d_in[0];   // [B,T,D]
    const float* h0 = (const float*)d_in[1];   // [B,D]
    const float* s0 = (const float*)d_in[2];   // [B,D]
    const float* W  = (const float*)d_in[3];   // [6D,D]
    const float* bv = (const float*)d_in[4];   // [6D]

    float* out   = (float*)d_out;                        // [B,T,D]
    float* s_out = out + (size_t)MM * DD;                // [B,D]

    char* ws = (char*)d_ws;
    size_t off = 0;
    unsigned short* xb = (unsigned short*)(ws + off); off += (size_t)MM * KK * 2;
    unsigned short* wb = (unsigned short*)(ws + off); off += (size_t)NG * DD * KK * 2;
    off = (off + 255) & ~(size_t)255;
    float4* scanbuf = (float4*)(ws + off);

    // 1) pack x and W to bf16
    {
        int nx = MM * KK;
        pack_bf16_kernel<<<8192, 256, 0, stream>>>(x, xb, nx);
        int nw = NG * DD * KK;
        pack_bf16_kernel<<<4096, 256, 0, stream>>>(W, wb, nw);
    }

    // 2) fused GEMM + RMS + gate nonlinearities
    {
        dim3 grid(DD / 16, MM / 256);   // 64 x 64 blocks, 8 waves each
        gates_gemm_kernel<<<grid, 256, 0, stream>>>(xb, wb, bv, scanbuf);
    }

    // 3) sequential scan over T
    {
        scan_kernel<<<(BB * DD) / 256, 256, 0, stream>>>(h0, s0, scanbuf, out, s_out);
    }
}
